// MultiheadFlashDiffV2_5308579578313
// MI455X (gfx1250) — matrix operand order repaired and checked by reading, not compiled, not run on hardware
//
#include <hip/hip_runtime.h>
#include <hip/hip_bf16.h>

typedef __bf16 bf16_t;
typedef __attribute__((ext_vector_type(16))) __bf16 v16bf;
typedef __attribute__((ext_vector_type(8)))  float  v8f;

#define BATCH  2
#define TLEN   2048
#define DMODEL 2048
#define NH     16
#define HQ     32
#define HKV    16
#define HD     128
#define SCALE  0.08838834764831845f   // 1/sqrt(128)

__device__ __forceinline__ bf16_t f2b(float f)  { return (bf16_t)f; }
__device__ __forceinline__ float  b2f(bf16_t b) { return (float)b; }

__device__ __forceinline__ v8f wmma_bf16(v16bf a, v16bf b, v8f c) {
  return __builtin_amdgcn_wmma_f32_16x16x32_bf16(false, a, false, b, (short)0, c, false, false);
}

union frag { v16bf v; uint4 u[2]; };

// ---------------------------------------------------------------- convert
__global__ void k_f32_to_bf16(const float* __restrict__ in,
                              bf16_t* __restrict__ out, int n) {
  int i = (blockIdx.x * blockDim.x + threadIdx.x) * 4;
  if (i + 3 < n) {
    float4 v = *(const float4*)(in + i);
    out[i]     = f2b(v.x); out[i + 1] = f2b(v.y);
    out[i + 2] = f2b(v.z); out[i + 3] = f2b(v.w);
  }
}

// W[K][N] f32 -> WT[N][K] bf16
__global__ __launch_bounds__(256)
void k_transpose_w(const float* __restrict__ W, bf16_t* __restrict__ WT,
                   int K, int N) {
  __shared__ float tile[32][33];
  int n0 = blockIdx.x * 32, k0 = blockIdx.y * 32;
  int id = threadIdx.x;
#pragma unroll
  for (int p = 0; p < 4; p++) {
    int kk = p * 8 + (id >> 5), nn = id & 31;
    tile[kk][nn] = W[(size_t)(k0 + kk) * N + n0 + nn];
  }
  __syncthreads();
#pragma unroll
  for (int p = 0; p < 4; p++) {
    int nn = p * 8 + (id >> 5), kk = id & 31;
    WT[(size_t)(n0 + nn) * K + k0 + kk] = f2b(tile[kk][nn]);
  }
}

// ---------------------------------------------------------------- GEMM
// C[M][N] = A[M][K](bf16,row-major) * B, with B given as BT[N][K] bf16.
// Block: 256 thr (8 waves); tile 256(M) x 64(N); wave strip 32(M) x 64(N).
// Double-buffered LDS B tile: 1 barrier / 32-k step, each B frag feeds 2 WMMAs.
__global__ __launch_bounds__(256)
void k_gemm_bf16(const bf16_t* __restrict__ A, const bf16_t* __restrict__ BT,
                 void* __restrict__ C, int M, int N, int K, int c_is_f32) {
  __shared__ __align__(16) bf16_t bt[2][64][32];   // [buf][n][k]
  const int tid  = threadIdx.x;
  const int lane = tid & 31;
  const int wave = tid >> 5;
  const int m0   = blockIdx.x * 256 + wave * 32;
  const int n0   = blockIdx.y * 64;
  const int half = lane >> 4;
  const int sn   = tid >> 2;            // staging row   (0..63)
  const int sk   = (tid & 3) * 8;       // staging k-off (0,8,16,24)

  v8f acc[2][4] = {{{}, {}, {}, {}}, {{}, {}, {}, {}}};

  // preload buffer 0
  *(uint4*)&bt[0][sn][sk] = *(const uint4*)&BT[(size_t)(n0 + sn) * K + sk];

  int buf = 0;
  for (int k0 = 0; k0 < K; k0 += 32) {
    __syncthreads();                    // bt[buf] staged; bt[buf^1] free
    if (k0 + 32 < K)
      *(uint4*)&bt[buf ^ 1][sn][sk] =
          *(const uint4*)&BT[(size_t)(n0 + sn) * K + k0 + 32 + sk];

    frag a0, a1;
    const bf16_t* ap0 = A + (size_t)(m0 + (lane & 15)) * K + k0 + half * 16;
    const bf16_t* ap1 = ap0 + (size_t)16 * K;
    a0.u[0] = *(const uint4*)ap0;  a0.u[1] = *(const uint4*)(ap0 + 8);
    a1.u[0] = *(const uint4*)ap1;  a1.u[1] = *(const uint4*)(ap1 + 8);
    __builtin_prefetch(ap0 + 32, 0, 1);     // next k-step A stream
    __builtin_prefetch(ap1 + 32, 0, 1);

#pragma unroll
    for (int nt = 0; nt < 4; nt++) {
      frag bq;
      const bf16_t* bp = &bt[buf][nt * 16 + (lane & 15)][half * 16];
      bq.u[0] = *(const uint4*)bp;
      bq.u[1] = *(const uint4*)(bp + 8);
      acc[0][nt] = wmma_bf16(a0.v, bq.v, acc[0][nt]);
      acc[1][nt] = wmma_bf16(a1.v, bq.v, acc[1][nt]);
    }
    buf ^= 1;
  }

  const int nl = lane & 15, mb = half * 8;
#pragma unroll
  for (int mh = 0; mh < 2; mh++)
#pragma unroll
    for (int nt = 0; nt < 4; nt++)
#pragma unroll
      for (int g = 0; g < 8; g++) {
        size_t idx = (size_t)(m0 + mh * 16 + mb + g) * N + n0 + nt * 16 + nl;
        if (c_is_f32) ((float*)C)[idx]  = acc[mh][nt][g];
        else          ((bf16_t*)C)[idx] = f2b(acc[mh][nt][g]);
      }
}

// ---------------------------------------------------------------- RoPE
// lin[b*T+t][h*HD + d] -> out[((b*nh+h)*T+t)*HD + d], interleaved rope
__global__ void k_rope(const bf16_t* __restrict__ lin,
                       const float* __restrict__ cosb,
                       const float* __restrict__ sinb,
                       bf16_t* __restrict__ outp, int nheads) {
  int idx = blockIdx.x;
  int t = idx % TLEN;
  int h = (idx / TLEN) % nheads;
  int b = idx / (TLEN * nheads);
  int i = threadIdx.x;                       // 0..63 pair index
  const bf16_t* src = lin + ((size_t)(b * TLEN + t)) * (nheads * HD) + h * HD + 2 * i;
  float x1 = b2f(src[0]), x2 = b2f(src[1]);
  float c = cosb[t * 64 + i], s = sinb[t * 64 + i];
  bf16_t* dst = outp + (((size_t)(b * nheads + h)) * TLEN + t) * HD + 2 * i;
  dst[0] = f2b(x1 * c - x2 * s);
  dst[1] = f2b(x1 * s + x2 * c);
}

// v_lin[b*T+t][h*HD+d] -> vt[((b*HKV+h)*HD + d)*T + t]
__global__ __launch_bounds__(256)
void k_vpack(const bf16_t* __restrict__ vlin, bf16_t* __restrict__ vt) {
  __shared__ bf16_t tile[32][HD + 2];
  int bh = blockIdx.x, t0 = blockIdx.y * 32;
  int h = bh & 15, b = bh >> 4;
  int id = threadIdx.x;
#pragma unroll
  for (int p = 0; p < 16; p++) {
    int tt = p * 2 + (id >> 7), d = id & 127;
    tile[tt][d] = vlin[((size_t)(b * TLEN + t0 + tt)) * (HKV * HD) + h * HD + d];
  }
  __syncthreads();
  bf16_t* base = vt + ((size_t)bh * HD) * TLEN;
#pragma unroll
  for (int p = 0; p < 16; p++) {
    int d = p * 8 + (id >> 5), tt = id & 31;
    base[(size_t)d * TLEN + t0 + tt] = tile[tt][d];
  }
}

// ---------------------------------------------------------------- lambda
__global__ __launch_bounds__(256)
void k_lambda(const float* __restrict__ x, const float* __restrict__ Wl,
              float* __restrict__ lam) {
  __shared__ float xr[DMODEL];
  __shared__ float red[16][16];
  int row = blockIdx.x, tid = threadIdx.x;
#pragma unroll
  for (int p = 0; p < 8; p++)
    xr[p * 256 + tid] = x[(size_t)row * DMODEL + p * 256 + tid];
  __syncthreads();
  int h = tid >> 4, gl = tid & 15;
  float acc = 0.f;
  for (int k = gl; k < DMODEL; k += 16) acc += xr[k] * Wl[(size_t)k * 16 + h];
  red[h][gl] = acc;
  __syncthreads();
  if (gl == 0) {
    float s = 0.f;
#pragma unroll
    for (int j = 0; j < 16; j++) s += red[h][j];
    lam[(size_t)row * 16 + h] = 1.f / (1.f + __expf(-s));
  }
}

// ---------------------------------------------------------------- flash attn
// grid(B*HQ, T/64), 128 threads = 4 waves, wave owns 16 q rows.
__global__ __launch_bounds__(128)
void k_flash_attn(const bf16_t* __restrict__ qh, const bf16_t* __restrict__ kh,
                  const bf16_t* __restrict__ vt, bf16_t* __restrict__ attn) {
  __shared__ float             stile[4][16][32];
  __shared__ __align__(16) bf16_t ptile[4][16][32];
  __shared__ float             rfac[4][16];
  __shared__ float             rinv[4][16];

  const int bh = blockIdx.x;
  const int b = bh >> 5, h = bh & 31, kvh = h >> 1;
  const int wave = threadIdx.x >> 5, lane = threadIdx.x & 31;
  const int r0 = blockIdx.y * 64 + wave * 16;
  const int m = lane & 15, half = lane >> 4;

  const bf16_t* qbase = qh + ((size_t)bh * TLEN + r0) * HD;
  const bf16_t* kbase = kh + (size_t)(b * HKV + kvh) * TLEN * HD;
  const bf16_t* vbase = vt + (size_t)(b * HKV + kvh) * HD * TLEN;

  frag aq[4];
#pragma unroll
  for (int kc = 0; kc < 4; kc++) {
    const bf16_t* p = qbase + (size_t)m * HD + kc * 32 + half * 16;
    aq[kc].u[0] = *(const uint4*)p;
    aq[kc].u[1] = *(const uint4*)(p + 8);
  }
  v8f o[8] = {{}, {}, {}, {}, {}, {}, {}, {}};
  float rmax = -3.4e38f, rsum = 0.f;

  const int nblk = ((r0 + 15) >> 5) + 1;
  for (int blk = 0; blk < nblk; blk++) {
    const int s0 = blk * 32;
    v8f sc0 = {}, sc1 = {};
#pragma unroll
    for (int kc = 0; kc < 4; kc++) {
      frag bk0, bk1;
      const bf16_t* p0 = kbase + (size_t)(s0 + m) * HD + kc * 32 + half * 16;
      const bf16_t* p1 = kbase + (size_t)(s0 + 16 + m) * HD + kc * 32 + half * 16;
      bk0.u[0] = *(const uint4*)p0;  bk0.u[1] = *(const uint4*)(p0 + 8);
      bk1.u[0] = *(const uint4*)p1;  bk1.u[1] = *(const uint4*)(p1 + 8);
      sc0 = wmma_bf16(aq[kc].v, bk0.v, sc0);
      sc1 = wmma_bf16(aq[kc].v, bk1.v, sc1);
    }
#pragma unroll
    for (int g = 0; g < 8; g++) {
      stile[wave][half * 8 + g][m]      = sc0[g];
      stile[wave][half * 8 + g][16 + m] = sc1[g];
    }
    asm volatile("s_wait_dscnt 0x0" ::: "memory");

    if (lane < 16) {                 // lane m owns q row r0+m
      const int r = r0 + m;
      float vals[32], vmaxb = -3.4e38f;
#pragma unroll
      for (int j = 0; j < 32; j++) {
        float sv = stile[wave][m][j] * SCALE;
        sv = ((s0 + j) <= r) ? sv : -3.4e38f;
        vals[j] = sv;
        vmaxb = fmaxf(vmaxb, sv);
      }
      float newmax = fmaxf(rmax, vmaxb);
      float fac = __expf(rmax - newmax);
      float psum = 0.f;
#pragma unroll
      for (int j = 0; j < 32; j++) {
        float p = __expf(vals[j] - newmax);
        psum += p;
        ptile[wave][m][j] = f2b(p);
      }
      rsum = rsum * fac + psum;
      rmax = newmax;
      rfac[wave][m] = fac;
    }
    asm volatile("s_wait_dscnt 0x0" ::: "memory");

#pragma unroll
    for (int g = 0; g < 8; g++) {    // rescale O
      float f = rfac[wave][half * 8 + g];
#pragma unroll
      for (int t = 0; t < 8; t++) o[t][g] *= f;
    }
    frag ap;
    {
      const bf16_t* p = &ptile[wave][m][half * 16];
      ap.u[0] = *(const uint4*)p;
      ap.u[1] = *(const uint4*)(p + 8);
    }
#pragma unroll
    for (int nt = 0; nt < 8; nt++) { // O += P * V
      frag bv;
      const bf16_t* p = vbase + (size_t)(nt * 16 + m) * TLEN + s0 + half * 16;
      bv.u[0] = *(const uint4*)p;  bv.u[1] = *(const uint4*)(p + 8);
      o[nt] = wmma_bf16(ap.v, bv.v, o[nt]);
    }
  }

  if (lane < 16) rinv[wave][m] = 1.f / rsum;
  asm volatile("s_wait_dscnt 0x0" ::: "memory");
  bf16_t* abase = attn + ((size_t)bh * TLEN + r0) * HD;
#pragma unroll
  for (int g = 0; g < 8; g++) {
    float inv = rinv[wave][half * 8 + g];
#pragma unroll
    for (int nt = 0; nt < 8; nt++)
      abase[(size_t)(half * 8 + g) * HD + nt * 16 + m] = f2b(o[nt][g] * inv);
  }
}

// ---------------------------------------------------------------- combine
__global__ void k_combine(const bf16_t* __restrict__ attn,
                          const float* __restrict__ lam,
                          bf16_t* __restrict__ z) {
  size_t i = (size_t)blockIdx.x * blockDim.x + threadIdx.x;  // B*T*NH*HD
  int d = i & 127;
  int h = (i >> 7) & 15;
  size_t bt = i >> 11;
  int t = bt & (TLEN - 1);
  int b = (int)(bt >> 11);
  float a1 = b2f(attn[(((size_t)(b * HQ + 2 * h))     * TLEN + t) * HD + d]);
  float a2 = b2f(attn[(((size_t)(b * HQ + 2 * h + 1)) * TLEN + t) * HD + d]);
  z[i] = f2b(a1 - lam[bt * 16 + h] * a2);
}

// ---------------------------------------------------------------- launch
extern "C" void kernel_launch(void* const* d_in, const int* in_sizes, int n_in,
                              void* d_out, int out_size, void* d_ws, size_t ws_size,
                              hipStream_t stream) {
  const float* x    = (const float*)d_in[0];
  const float* cosb = (const float*)d_in[1];
  const float* sinb = (const float*)d_in[2];
  const float* Wq   = (const float*)d_in[3];
  const float* Wk   = (const float*)d_in[4];
  const float* Wv   = (const float*)d_in[5];
  const float* Wo   = (const float*)d_in[6];
  const float* Wlam = (const float*)d_in[7];
  float* out = (float*)d_out;

  char* wsp = (char*)d_ws;
  auto alloc = [&](size_t bytes) -> void* {
    void* p = wsp; wsp += (bytes + 255) & ~(size_t)255; return p;
  };
  const int M = BATCH * TLEN;                       // 4096
  bf16_t* xb    = (bf16_t*)alloc((size_t)M * DMODEL * 2);
  bf16_t* WqT   = (bf16_t*)alloc((size_t)HQ * HD * DMODEL * 2);
  bf16_t* WkT   = (bf16_t*)alloc((size_t)HKV * HD * DMODEL * 2);
  bf16_t* WvT   = (bf16_t*)alloc((size_t)HKV * HD * DMODEL * 2);
  bf16_t* WoT   = (bf16_t*)alloc((size_t)DMODEL * DMODEL * 2);
  bf16_t* q_lin = (bf16_t*)alloc((size_t)M * HQ * HD * 2);
  bf16_t* k_lin = (bf16_t*)alloc((size_t)M * HKV * HD * 2);
  bf16_t* v_lin = (bf16_t*)alloc((size_t)M * HKV * HD * 2);
  bf16_t* qhb   = (bf16_t*)alloc((size_t)M * HQ * HD * 2);
  bf16_t* khb   = (bf16_t*)alloc((size_t)M * HKV * HD * 2);
  bf16_t* vtb   = (bf16_t*)alloc((size_t)M * HKV * HD * 2);
  float*  lamb  = (float*) alloc((size_t)M * NH * 4);
  bf16_t* attnb = (bf16_t*)alloc((size_t)M * HQ * HD * 2);
  bf16_t* zb    = (bf16_t*)alloc((size_t)M * DMODEL * 2);

  int nx = M * DMODEL;
  k_f32_to_bf16<<<nx / 1024, 256, 0, stream>>>(x, xb, nx);

  k_transpose_w<<<dim3(HQ * HD / 32, DMODEL / 32), 256, 0, stream>>>(Wq, WqT, DMODEL, HQ * HD);
  k_transpose_w<<<dim3(HKV * HD / 32, DMODEL / 32), 256, 0, stream>>>(Wk, WkT, DMODEL, HKV * HD);
  k_transpose_w<<<dim3(HKV * HD / 32, DMODEL / 32), 256, 0, stream>>>(Wv, WvT, DMODEL, HKV * HD);
  k_transpose_w<<<dim3(DMODEL / 32, DMODEL / 32), 256, 0, stream>>>(Wo, WoT, DMODEL, DMODEL);

  k_gemm_bf16<<<dim3(M / 256, HQ * HD / 64), 256, 0, stream>>>(xb, WqT, q_lin, M, HQ * HD, DMODEL, 0);
  k_gemm_bf16<<<dim3(M / 256, HKV * HD / 64), 256, 0, stream>>>(xb, WkT, k_lin, M, HKV * HD, DMODEL, 0);
  k_gemm_bf16<<<dim3(M / 256, HKV * HD / 64), 256, 0, stream>>>(xb, WvT, v_lin, M, HKV * HD, DMODEL, 0);

  k_rope<<<BATCH * HQ * TLEN, 64, 0, stream>>>(q_lin, cosb, sinb, qhb, HQ);
  k_rope<<<BATCH * HKV * TLEN, 64, 0, stream>>>(k_lin, cosb, sinb, khb, HKV);
  k_vpack<<<dim3(BATCH * HKV, TLEN / 32), 256, 0, stream>>>(v_lin, vtb);

  k_lambda<<<M, 256, 0, stream>>>(x, Wlam, lamb);

  k_flash_attn<<<dim3(BATCH * HQ, TLEN / 64), 128, 0, stream>>>(qhb, khb, vtb, attnb);

  k_combine<<<(M * DMODEL) / 256, 256, 0, stream>>>(attnb, lamb, zb);

  k_gemm_bf16<<<dim3(M / 256, DMODEL / 64), 256, 0, stream>>>(zb, WoT, out, M, DMODEL, DMODEL, 1);
}
